// TGC_60842506715220
// MI455X (gfx1250) — matrix operand order repaired and checked by reading, not compiled, not run on hardware
//
#include <hip/hip_runtime.h>
#include <hip/hip_bf16.h>

typedef __bf16 v16bf __attribute__((ext_vector_type(16)));
typedef __bf16 v8bf  __attribute__((ext_vector_type(8)));
typedef float  v8f   __attribute__((ext_vector_type(8)));

#define TSTEPS 20
#define FIN 6
#define HDIM 64

__device__ __forceinline__ float fsig(float x)  { return 1.0f / (1.0f + __expf(-x)); }
__device__ __forceinline__ float ftanh(float x) { float e = __expf(2.0f * x); return 1.0f - 2.0f / (e + 1.0f); }

__device__ __forceinline__ v8f wmma_bf16(v16bf a, v16bf b, v8f c) {
    return __builtin_amdgcn_wmma_f32_16x16x32_bf16(false, a, false, b, (short)0, c, false, false);
}

// order-preserving float <-> uint encoding for atomic segment-max (key 0 == "-inf, never written")
__device__ __forceinline__ unsigned enc_f(float f) {
    unsigned u = __float_as_uint(f);
    return (u & 0x80000000u) ? ~u : (u | 0x80000000u);
}
__device__ __forceinline__ float dec_key(unsigned k) {
    return (k & 0x80000000u) ? __uint_as_float(k ^ 0x80000000u) : __uint_as_float(~k);
}

// ---------------------------------------------------------------------------
// K1: LSTM over T steps. 4 waves/block, 16 nodes/wave -> 64 nodes/block.
// gates(16x256) = [h | x | 1] @ [W_hh.T ; W_ih.T ; bias] with zero-init C
// (bias folded in as a constant-1 column at k==6 of the padded x chunk, so
//  every tile's WMMA chain starts from the inline-0 SRC2 form)
// ---------------------------------------------------------------------------
__global__ __launch_bounds__(128) void lstm_kernel(
    const float* __restrict__ x, const float* __restrict__ W_ih,
    const float* __restrict__ W_hh, const float* __restrict__ b_ih,
    const float* __restrict__ b_hh, float* __restrict__ h_out, int N)
{
    __shared__ __bf16 sWhh[256 * 64];   // [n][k] = W_hh[n][k]  (bf16) 32KB
    __shared__ __bf16 sWih[256 * 32];   // [n][k]: k<6 = W_ih[n][k], k==6 = bias[n], else 0
    __shared__ __bf16 sH[64 * 64];      // h (bf16), 64 nodes     8KB

    const int tid  = threadIdx.x;
    const int wave = tid >> 5;
    const int lane = tid & 31;
    const int lo   = lane & 15;
    const int hi   = lane >> 4;

    for (int i = tid; i < 256 * 64; i += 128) sWhh[i] = (__bf16)W_hh[i];
    for (int i = tid; i < 256 * 32; i += 128) {
        int n = i >> 5, k = i & 31;
        __bf16 v = (__bf16)0.0f;
        if (k < FIN)       v = (__bf16)W_ih[n * FIN + k];
        else if (k == FIN) v = (__bf16)(b_ih[n] + b_hh[n]);   // bias row, pairs with A's 1.0
        sWih[i] = v;
    }
    for (int i = tid; i < 64 * 64; i += 128) sH[i] = (__bf16)0.0f;
    __syncthreads();

    const int nodeBase = blockIdx.x * 64 + wave * 16;
    const __bf16* myH = sH + (wave * 16) * HDIM;

    v8f   acc[16];
    float cst[4][8];
#pragma unroll
    for (int nb = 0; nb < 4; ++nb)
#pragma unroll
        for (int r = 0; r < 8; ++r) cst[nb][r] = 0.0f;

    for (int t = 0; t < TSTEPS; ++t) {
        // ---- A fragments (contiguous k order, same as B: lane<16 K=0..15, lane>=16 K=16..31) ----
        v16bf a0, a1, ax;
        {
            const __bf16* p = myH + lo * HDIM;
            const int kb = hi ? 16 : 0;
#pragma unroll
            for (int i = 0; i < 8; ++i) {
                a0[i]     = p[kb + i];        a0[i + 8] = p[kb + 8 + i];
                a1[i]     = p[32 + kb + i];   a1[i + 8] = p[32 + kb + 8 + i];
            }
#pragma unroll
            for (int i = 0; i < 16; ++i) ax[i] = (__bf16)0.0f;
            if (!hi) {
                ax[FIN] = (__bf16)1.0f;       // constant-1 column -> bias row of B
                const int node = nodeBase + lo;
                if (node < N) {
                    const float* xp = x + ((size_t)node * TSTEPS + t) * FIN;
#pragma unroll
                    for (int k = 0; k < FIN; ++k) ax[k] = (__bf16)xp[k];
                }
            }
        }
        // ---- 16 N-tiles x 3 K-chunks of WMMA, zero-init C ----
#pragma unroll
        for (int nb = 0; nb < 16; ++nb) {
            const int n   = nb * 16 + lo;
            const int kbB = hi ? 16 : 0;
            v8f c = {};
            v16bf b0 = *(const v16bf*)(sWhh + n * 64 + kbB);
            v16bf b1 = *(const v16bf*)(sWhh + n * 64 + 32 + kbB);
            v16bf bx = *(const v16bf*)(sWih + n * 32 + kbB);
            c = wmma_bf16(ax, bx, c);
            c = wmma_bf16(a0, b0, c);
            c = wmma_bf16(a1, b1, c);
            acc[nb] = c;
        }
        // ---- cell update: i/f/g/o for (m,j) all live in this lane ----
#pragma unroll
        for (int nb = 0; nb < 4; ++nb) {
#pragma unroll
            for (int r = 0; r < 8; ++r) {
                const float ig = fsig(acc[nb][r]);
                const float fg = fsig(acc[nb + 4][r]);
                const float gg = ftanh(acc[nb + 8][r]);
                const float og = fsig(acc[nb + 12][r]);
                float c = fg * cst[nb][r] + ig * gg;
                cst[nb][r] = c;
                const float h = og * ftanh(c);
                const int m = r + 8 * hi;
                const int j = nb * 16 + lo;
                sH[(wave * 16 + m) * HDIM + j] = (__bf16)h;
                if (t == TSTEPS - 1) {
                    const int node = nodeBase + m;
                    if (node < N) h_out[(size_t)node * HDIM + j] = h;
                }
            }
        }
        __syncthreads();
    }
}

// ---------------------------------------------------------------------------
// K2: hw = h @ W_conv (WMMA bf16), s_node/d_node dots, init gout/segmax/denom
// ---------------------------------------------------------------------------
__global__ __launch_bounds__(128) void node_prep_kernel(
    const float* __restrict__ h, const float* __restrict__ W_conv,
    const float* __restrict__ att_src, const float* __restrict__ att_dst,
    const float* __restrict__ conv_bias,
    float* __restrict__ hw, float* __restrict__ s_node, float* __restrict__ d_node,
    unsigned* __restrict__ segmax, float* __restrict__ denom,
    float* __restrict__ gout, int N)
{
    __shared__ __bf16 sWcT[64 * 64];  // [n][k] = W_conv[k][n]
    __shared__ __bf16 sHn[64 * 64];
    __shared__ float  shw[64 * 64];

    const int tid = threadIdx.x, wave = tid >> 5, lane = tid & 31;
    const int lo = lane & 15, hi = lane >> 4;
    const int nodeBase = blockIdx.x * 64;

    for (int i = tid; i < 64 * 64; i += 128) {
        const int k = i >> 6, n = i & 63;
        sWcT[n * 64 + k] = (__bf16)W_conv[i];
        const int node = nodeBase + k;
        sHn[i] = (__bf16)((node < N) ? h[(size_t)node * HDIM + n] : 0.0f);
    }
    __syncthreads();

    const int mwave = wave * 16;
    v16bf a0, a1;
    {
        const __bf16* p = sHn + (mwave + lo) * HDIM;
        const int kb = hi ? 16 : 0;
#pragma unroll
        for (int i = 0; i < 8; ++i) {
            a0[i]     = p[kb + i];        a0[i + 8] = p[kb + 8 + i];
            a1[i]     = p[32 + kb + i];   a1[i + 8] = p[32 + kb + 8 + i];
        }
    }
#pragma unroll
    for (int nb = 0; nb < 4; ++nb) {
        const int n   = nb * 16 + lo;
        const int kbB = hi ? 16 : 0;
        v8f c = {};
        v16bf b0 = *(const v16bf*)(sWcT + n * 64 + kbB);
        v16bf b1 = *(const v16bf*)(sWcT + n * 64 + 32 + kbB);
        c = wmma_bf16(a0, b0, c);
        c = wmma_bf16(a1, b1, c);
        const float cb = conv_bias[n];
#pragma unroll
        for (int r = 0; r < 8; ++r) {
            const int m = r + 8 * hi;
            const int node = nodeBase + mwave + m;
            shw[(mwave + m) * HDIM + n] = c[r];
            if (node < N) {
                hw[(size_t)node * HDIM + n]   = c[r];
                gout[(size_t)node * HDIM + n] = cb;   // out = segsum(msg) + conv_bias
            }
        }
    }
    __syncthreads();

    // lanes 0-15: s_node for node (mwave+lo); lanes 16-31: d_node
    const int nl = mwave + lo;
    const int node = nodeBase + nl;
    const float* av = hi ? att_dst : att_src;
    float s = 0.0f;
#pragma unroll 8
    for (int j = 0; j < HDIM; ++j) s += shw[nl * HDIM + j] * av[j];
    if (node < N) {
        if (hi) d_node[node] = s;
        else { s_node[node] = s; segmax[node] = 0u; denom[node] = 0.0f; }
    }
}

// ---------------------------------------------------------------------------
// Edge pipeline (memory bound; node arrays are L2-resident at 192MB L2)
// ---------------------------------------------------------------------------
__global__ void prep_wea_kernel(const float* __restrict__ W_edge,
                                const float* __restrict__ att_edge,
                                float* __restrict__ wea)
{
    const int k = threadIdx.x;
    if (k < 4) {
        float s = 0.0f;
        for (int j = 0; j < HDIM; ++j) s += W_edge[k * HDIM + j] * att_edge[j];
        wea[k] = s;
    }
}

__global__ __launch_bounds__(256) void edge_logits_kernel(
    const int* __restrict__ ei, const float* __restrict__ ea,
    const float* __restrict__ sn, const float* __restrict__ dn,
    const float* __restrict__ wea, float* __restrict__ logit,
    unsigned* __restrict__ segmax, int E)
{
    const int e = blockIdx.x * 256 + threadIdx.x;
    if (e >= E) return;
    const int src = ei[e], dst = ei[E + e];
    const float4 a4 = ((const float4*)ea)[e];
    float l = sn[src] + dn[dst] + a4.x * wea[0] + a4.y * wea[1] + a4.z * wea[2] + a4.w * wea[3];
    l = (l > 0.0f) ? l : 0.2f * l;          // leaky_relu(0.2)
    logit[e] = l;
    atomicMax(&segmax[dst], enc_f(l));
}

__global__ __launch_bounds__(256) void edge_exp_kernel(
    const int* __restrict__ ei, float* __restrict__ logit,
    const unsigned* __restrict__ segmax, float* __restrict__ denom, int E)
{
    const int e = blockIdx.x * 256 + threadIdx.x;
    if (e >= E) return;
    const int dst = ei[E + e];
    const unsigned key = segmax[dst];
    const float sm = (key == 0u) ? 0.0f : dec_key(key);   // isfinite -> 0 for no-edge nodes
    const float ex = __expf(logit[e] - sm);
    logit[e] = ex;
    atomicAdd(&denom[dst], ex);
}

// one wave per edge: gather hw[src] as float2/lane, scatter-add alpha*hw into gout[dst]
__global__ __launch_bounds__(256) void edge_scatter_kernel(
    const int* __restrict__ ei, const float* __restrict__ exbuf,
    const float* __restrict__ denom, const float* __restrict__ hw,
    float* __restrict__ gout, int E)
{
    const int e = blockIdx.x * 8 + (threadIdx.x >> 5);
    if (e >= E) return;
    const int lane = threadIdx.x & 31;
    const int src = ei[e], dst = ei[E + e];
    const float alpha = exbuf[e] / (denom[dst] + 1e-16f);
    const float2 v = ((const float2*)(hw + (size_t)src * HDIM))[lane];
    float* gp = gout + (size_t)dst * HDIM + lane * 2;
    atomicAdd(gp, alpha * v.x);
    atomicAdd(gp + 1, alpha * v.y);
}

// ---------------------------------------------------------------------------
// K4: out = sigmoid([h|g] @ out_W.T + out_b)
// ---------------------------------------------------------------------------
__global__ __launch_bounds__(256) void out_head_kernel(
    const float* __restrict__ h, const float* __restrict__ g,
    const float* __restrict__ W, const float* __restrict__ b,
    float* __restrict__ out, int N)
{
    const int n = blockIdx.x * 256 + threadIdx.x;
    if (n >= N) return;
    float acc[4] = { b[0], b[1], b[2], b[3] };
    const float* hp = h + (size_t)n * HDIM;
    const float* gp = g + (size_t)n * HDIM;
#pragma unroll 4
    for (int j = 0; j < HDIM; ++j) {
        const float hv = hp[j];
#pragma unroll
        for (int r = 0; r < 4; ++r) acc[r] += W[r * 128 + j] * hv;
    }
#pragma unroll 4
    for (int j = 0; j < HDIM; ++j) {
        const float gv = gp[j];
#pragma unroll
        for (int r = 0; r < 4; ++r) acc[r] += W[r * 128 + 64 + j] * gv;
    }
#pragma unroll
    for (int r = 0; r < 4; ++r) out[(size_t)n * 4 + r] = fsig(acc[r]);
}

// ---------------------------------------------------------------------------
extern "C" void kernel_launch(void* const* d_in, const int* in_sizes, int n_in,
                              void* d_out, int out_size, void* d_ws, size_t ws_size,
                              hipStream_t stream)
{
    const float* x        = (const float*)d_in[0];
    const int*   ei       = (const int*)  d_in[1];
    const float* ea       = (const float*)d_in[2];
    const float* W_ih     = (const float*)d_in[3];
    const float* W_hh     = (const float*)d_in[4];
    const float* b_ih     = (const float*)d_in[5];
    const float* b_hh     = (const float*)d_in[6];
    const float* W_conv   = (const float*)d_in[7];
    const float* W_edge   = (const float*)d_in[8];
    const float* att_src  = (const float*)d_in[9];
    const float* att_dst  = (const float*)d_in[10];
    const float* att_edge = (const float*)d_in[11];
    const float* conv_b   = (const float*)d_in[12];
    const float* out_W    = (const float*)d_in[13];
    const float* out_b    = (const float*)d_in[14];

    const int N = in_sizes[0] / (TSTEPS * FIN);
    const int E = in_sizes[1] / 2;

    char* wsp = (char*)d_ws;
    auto carve = [&](size_t bytes) -> char* {
        char* p = wsp;
        wsp += (bytes + 255) & ~(size_t)255;
        return p;
    };
    float*    h    = (float*)   carve((size_t)N * HDIM * 4);
    float*    hw   = (float*)   carve((size_t)N * HDIM * 4);
    float*    gout = (float*)   carve((size_t)N * HDIM * 4);
    float*    sn   = (float*)   carve((size_t)N * 4);
    float*    dn   = (float*)   carve((size_t)N * 4);
    unsigned* smax = (unsigned*)carve((size_t)N * 4);
    float*    den  = (float*)   carve((size_t)N * 4);
    float*    lg   = (float*)   carve((size_t)E * 4);
    float*    wea  = (float*)   carve(256);
    (void)ws_size; (void)n_in; (void)out_size;

    prep_wea_kernel<<<1, 64, 0, stream>>>(W_edge, att_edge, wea);
    lstm_kernel<<<(N + 63) / 64, 128, 0, stream>>>(x, W_ih, W_hh, b_ih, b_hh, h, N);
    node_prep_kernel<<<(N + 63) / 64, 128, 0, stream>>>(h, W_conv, att_src, att_dst,
                                                        conv_b, hw, sn, dn, smax, den, gout, N);
    edge_logits_kernel<<<(E + 255) / 256, 256, 0, stream>>>(ei, ea, sn, dn, wea, lg, smax, E);
    edge_exp_kernel<<<(E + 255) / 256, 256, 0, stream>>>(ei, lg, smax, den, E);
    edge_scatter_kernel<<<(E + 7) / 8, 256, 0, stream>>>(ei, lg, den, hw, gout, E);
    out_head_kernel<<<(N + 255) / 256, 256, 0, stream>>>(h, gout, out_W, out_b,
                                                         (float*)d_out, N);
}
